// SwitchMoE_HS_22273700397479
// MI455X (gfx1250) — compile-verified, matrix operand order checked by reading, not run on hardware
//
#include <hip/hip_runtime.h>
#include <hip/hip_bf16.h>

// ---------------------------------------------------------------------------
// SwitchMoE + 2D selective scan (VMamba-style) for MI455X / gfx1250, wave32.
//
// GEMMs (x@Win 192->768, y@Wout 384->192) run on the matrix pipe with
// v_wmma_f32_16x16x32_bf16 (bf16 in, f32 accumulate), TN tiles per wave,
// A/B staged through LDS with GLOBAL_LOAD_ASYNC_TO_LDS_B128 (ASYNCcnt path).
// bf16 fragments are packed two-at-a-time with v_perm_b32 (1 op / 2 elems).
// The selective scan keeps 16 states/lane in VGPRs and broadcasts B/C rows
// via LDS. Problem is HBM/latency bound (~9 GFLOP, ~150 MB traffic vs
// 23.3 TB/s), so the GEMM goal is coalesced streams + matrix pipe, not peak.
// ---------------------------------------------------------------------------

#define BSZ   2
#define HHH   48
#define WWW   48
#define LTOK  (HHH * WWW)      // 2304
#define BL    (BSZ * LTOK)     // 4608
#define DM    192
#define NEXP  4
#define DST   16
#define DIN   384
#define DTR   12
#define NDBL  (DTR + 2 * DST)  // 44
#define MIDC  24

typedef __attribute__((ext_vector_type(16))) __bf16       v16bf;
typedef __attribute__((ext_vector_type(8)))  float        v8f;
typedef __attribute__((ext_vector_type(8)))  unsigned int v8u;
typedef __attribute__((ext_vector_type(4)))  int          v4i;

#define AS1 __attribute__((address_space(1)))
#define AS3 __attribute__((address_space(3)))

#if defined(__has_builtin)
#if __has_builtin(__builtin_amdgcn_global_load_async_to_lds_b128)
#define HAVE_ASYNC_LDS 1
#endif
#endif

__device__ __forceinline__ void wait_async_and_barrier() {
#if defined(HAVE_ASYNC_LDS)
#if __has_builtin(__builtin_amdgcn_s_wait_asynccnt)
  __builtin_amdgcn_s_wait_asynccnt(0);
#else
  asm volatile("s_wait_asynccnt 0" ::: "memory");
#endif
#endif
  __syncthreads();
}

// Pack two f32 -> two truncated bf16 in one dword: {hi[31:16], lo[31:16]}.
// Lowers to a single v_perm_b32.
__device__ __forceinline__ unsigned int pk2bf(float lo, float hi) {
#if __has_builtin(__builtin_amdgcn_perm)
  return __builtin_amdgcn_perm(__builtin_bit_cast(unsigned int, hi),
                               __builtin_bit_cast(unsigned int, lo),
                               0x07060302u);
#else
  unsigned int ul = __builtin_bit_cast(unsigned int, lo);
  unsigned int uh = __builtin_bit_cast(unsigned int, hi);
  return (ul >> 16) | (uh & 0xFFFF0000u);
#endif
}

__device__ __forceinline__ float siluf(float v) {
  return v / (1.f + __expf(-v));
}

// Snake serialization orders (matches make_orders with H=W=48, both even).
__device__ __forceinline__ int order_tok(int e, int t) {
  int r, c, tok;
  if (e == 0) {        // h-snake starting at bottom
    r = t / WWW; c = t - r * WWW;
    if (r & 1) c = WWW - 1 - c;
    tok = (HHH - 1 - r) * WWW + c;
  } else if (e == 1) { // h-snake starting at top
    r = t / WWW; c = t - r * WWW;
    if (r & 1) c = WWW - 1 - c;
    tok = r * WWW + c;
  } else if (e == 2) { // v-snake starting at left
    r = t / HHH; c = t - r * HHH;
    if (r & 1) c = HHH - 1 - c;
    tok = c * WWW + r;
  } else {             // v-snake starting at right
    r = t / HHH; c = t - r * HHH;
    if (r & 1) c = HHH - 1 - c;
    tok = c * WWW + (WWW - 1 - r);
  }
  return tok;
}

// ---------------------------------------------------------------------------
__global__ void k_zero(float* __restrict__ p, int n) {
  int i = blockIdx.x * blockDim.x + threadIdx.x;
  if (i < n) p[i] = 0.f;
}

// Router: logits = x @ Wg + bg; top-1 gate = softmax prob of argmax.
__global__ void k_router(const float* __restrict__ x, const float* __restrict__ Wg,
                         const float* __restrict__ bg, float* __restrict__ gates,
                         int* __restrict__ eidx) {
  int t = blockIdx.x * blockDim.x + threadIdx.x;
  if (t >= BL) return;
  const float* xr = x + (size_t)t * DM;
  float lg[NEXP];
  #pragma unroll
  for (int e = 0; e < NEXP; ++e) lg[e] = bg[e];
  for (int d = 0; d < DM; ++d) {
    float xv = xr[d];
    #pragma unroll
    for (int e = 0; e < NEXP; ++e) lg[e] += xv * Wg[d * NEXP + e];
  }
  int am = 0; float mx = lg[0];
  #pragma unroll
  for (int e = 1; e < NEXP; ++e) if (lg[e] > mx) { mx = lg[e]; am = e; }
  float s = 0.f;
  #pragma unroll
  for (int e = 0; e < NEXP; ++e) s += __expf(lg[e] - mx);
  gates[t] = 1.f / s;   // prob of the argmax expert
  eidx[t]  = am;
}

// ---------------------------------------------------------------------------
// WMMA GEMM: C[M x N] = A[M x K] @ B[K x N], row-major fp32 in memory,
// bf16 fragments, f32 accumulate.
//  - block = 128 threads = 4 waves; block tile = 16 rows x NB cols,
//    NB = 4*TN*16; each wave owns TN adjacent 16x16 C tiles.
//  - A (16 x K) staged to LDS once; B staged per 32-deep K chunk, both via
//    GLOBAL_LOAD_ASYNC_TO_LDS_B128 (signature: (v4i AS1*, v4i AS3*, imm, imm)).
//  - A fragment built once per chunk, reused by all TN WMMAs; fragments are
//    packed pairwise with v_perm_b32 (element pairs (2j,2j+1) are consecutive
//    K values per the 16-bit A/B VGPR layouts of ISA 7.12.2).
// If `gates` != null:  C[row,col] += gate_mask(row) * tile  (expert gating).
// ---------------------------------------------------------------------------
template <int K, int TN>
__global__ __launch_bounds__(128)
void k_wmma_gemm(const float* __restrict__ A, const float* __restrict__ Bm,
                 float* __restrict__ C, int ldb, int ldc,
                 const float* __restrict__ gates, const int* __restrict__ eidx,
                 int e) {
  constexpr int NB = 4 * TN * 16;
  __shared__ float sA[16 * K];
  __shared__ float sB[32 * NB];
  const int tid  = threadIdx.x;
  const int lane = tid & 31;
  const int wave = tid >> 5;
  const int m0   = blockIdx.x * 16;
  const int nblk = blockIdx.y * NB;

  // ---- stage A tile (16 x K) into LDS ----
#if defined(HAVE_ASYNC_LDS)
  for (int i = tid * 4; i < 16 * K; i += 128 * 4) {
    int r = i / K, c = i - r * K;                       // 16B-aligned (K%4==0)
    __builtin_amdgcn_global_load_async_to_lds_b128(
        (AS1 v4i*)(A + (size_t)(m0 + r) * K + c), (AS3 v4i*)&sA[i], 0, 0);
  }
#else
  for (int i = tid; i < 16 * K; i += 128) {
    int r = i / K, c = i - r * K;
    sA[i] = A[(size_t)(m0 + r) * K + c];
  }
#endif

  v8f acc[TN];
  #pragma unroll
  for (int tt = 0; tt < TN; ++tt) acc[tt] = {};

  const int lh = lane >> 4;        // lane half (0/1)
  const int nl = lane & 15;
  const int kb = lh << 3;          // K sub-offset within chunk

  #pragma unroll 2
  for (int k0 = 0; k0 < K; k0 += 32) {
    // ---- stage B chunk (32 x NB), coalesced along N ----
#if defined(HAVE_ASYNC_LDS)
    for (int i = tid * 4; i < 32 * NB; i += 128 * 4) {
      int kk = i / NB, nn = i - kk * NB;                // NB%4==0 -> aligned
      __builtin_amdgcn_global_load_async_to_lds_b128(
          (AS1 v4i*)(Bm + (size_t)(k0 + kk) * ldb + nblk + nn),
          (AS3 v4i*)&sB[i], 0, 0);
    }
#else
    for (int i = tid; i < 32 * NB; i += 128) {
      int kk = i / NB, nn = i - kk * NB;
      sB[i] = Bm[(size_t)(k0 + kk) * ldb + nblk + nn];
    }
#endif
    wait_async_and_barrier();

    // ---- build A fragment (shared by all TN tiles) ----
    // v16bf element pair (2j, 2j+1) <-> K = kb + 2j (+8 for upper half).
    v8u apk;
    const float* arow = &sA[nl * K + k0];
    #pragma unroll
    for (int jj = 0; jj < 8; ++jj) {
      int kk = kb + 2 * jj + (jj >= 4 ? 8 : 0);
      apk[jj] = pk2bf(arow[kk], arow[kk + 1]);
    }
    const v16bf af = __builtin_bit_cast(v16bf, apk);

    // ---- TN WMMAs on this chunk ----
    #pragma unroll
    for (int tt = 0; tt < TN; ++tt) {
      v8u bpk;
      const float* bcol = &sB[(wave * TN + tt) * 16 + nl];
      #pragma unroll
      for (int jj = 0; jj < 8; ++jj) {
        int kk = kb + 2 * jj + (jj >= 4 ? 8 : 0);
        bpk[jj] = pk2bf(bcol[(size_t)kk * NB], bcol[(size_t)(kk + 1) * NB]);
      }
      const v16bf bfv = __builtin_bit_cast(v16bf, bpk);
      acc[tt] = __builtin_amdgcn_wmma_f32_16x16x32_bf16(
          false, af, false, bfv, (short)0, acc[tt], false, false);
    }
    __syncthreads();   // protect sB before next chunk's staging
  }

  // ---- store: C/D layout VGPR v -> row v (+8 for upper lane half) ----
  const int rbase = m0 + (lh << 3);
  #pragma unroll
  for (int tt = 0; tt < TN; ++tt) {
    const int ncol = nblk + (wave * TN + tt) * 16 + nl;
    if (gates) {
      #pragma unroll
      for (int v = 0; v < 8; ++v) {
        int row = rbase + v;
        float g = (eidx[row] == e) ? gates[row] : 0.f;
        C[(size_t)row * ldc + ncol] += g * acc[tt][v];
      }
    } else {
      #pragma unroll
      for (int v = 0; v < 8; ++v)
        C[(size_t)(rbase + v) * ldc + ncol] = acc[tt][v];
    }
  }
}

// ---------------------------------------------------------------------------
// Conv path: pw1 (384->24, +bias), 3x3 depthwise SAME, pw2 (24->384) + SiLU.
// ---------------------------------------------------------------------------
__global__ void k_pw1(const float* __restrict__ xz, const float* __restrict__ w,
                      const float* __restrict__ b, float* __restrict__ hmid) {
  int i = blockIdx.x * blockDim.x + threadIdx.x;
  if (i >= BL * MIDC) return;
  int t = i / MIDC, m = i - t * MIDC;
  const float* xin = xz + (size_t)t * (2 * DIN);   // first half = xin
  float s = b[m];
  for (int c = 0; c < DIN; ++c) s += xin[c] * w[c * MIDC + m];
  hmid[i] = s;
}

__global__ void k_dwconv(const float* __restrict__ hin, const float* __restrict__ w,
                         float* __restrict__ hout) {
  int i = blockIdx.x * blockDim.x + threadIdx.x;
  if (i >= BL * MIDC) return;
  int t = i / MIDC, m = i - t * MIDC;
  int b = t / LTOK, p = t - b * LTOK;
  int h = p / WWW, wc = p - h * WWW;
  float s = 0.f;
  #pragma unroll
  for (int kh = 0; kh < 3; ++kh) {
    int h2 = h + kh - 1;
    if (h2 < 0 || h2 >= HHH) continue;
    #pragma unroll
    for (int kw = 0; kw < 3; ++kw) {
      int w2 = wc + kw - 1;
      if (w2 < 0 || w2 >= WWW) continue;
      s += hin[(size_t)(b * LTOK + h2 * WWW + w2) * MIDC + m] *
           w[(kh * 3 + kw) * MIDC + m];
    }
  }
  hout[i] = s;
}

__global__ void k_pw2silu(const float* __restrict__ hmid, const float* __restrict__ w,
                          float* __restrict__ xconv) {
  int i = blockIdx.x * blockDim.x + threadIdx.x;
  if (i >= BL * DIN) return;
  int t = i / DIN, c = i - t * DIN;
  float s = 0.f;
  #pragma unroll
  for (int m = 0; m < MIDC; ++m) s += hmid[t * MIDC + m] * w[m * DIN + c];
  xconv[i] = siluf(s);
}

// x_dbl = x_conv @ Wx  (384 -> 44: [dt | B | C])
__global__ void k_xdbl(const float* __restrict__ xconv, const float* __restrict__ Wx,
                       float* __restrict__ xdb) {
  int i = blockIdx.x * blockDim.x + threadIdx.x;
  if (i >= BL * NDBL) return;
  int t = i / NDBL, j = i - t * NDBL;
  float s = 0.f;
  for (int c = 0; c < DIN; ++c) s += xconv[(size_t)t * DIN + c] * Wx[c * NDBL + j];
  xdb[i] = s;
}

// delta = softplus(dtl @ Wdt + 2*bdt)
__global__ void k_delta(const float* __restrict__ xdb, const float* __restrict__ Wdt,
                        const float* __restrict__ bdt, float* __restrict__ dlt) {
  int i = blockIdx.x * blockDim.x + threadIdx.x;
  if (i >= BL * DIN) return;
  int t = i / DIN, c = i - t * DIN;
  float s = 2.f * bdt[c];
  #pragma unroll
  for (int r = 0; r < DTR; ++r) s += xdb[(size_t)t * NDBL + r] * Wdt[r * DIN + c];
  dlt[i] = (s > 20.f) ? s : log1pf(__expf(s));
}

// ---------------------------------------------------------------------------
// Selective scan. One lane per channel (16 states in VGPRs), B/C rows staged
// through LDS in 32-step chunks. delta/B/C are in original token order; only
// u is read through the snake order, and output scatters through it.
// ---------------------------------------------------------------------------
__global__ __launch_bounds__(128)
void k_scan(const float* __restrict__ dlt, const float* __restrict__ xdb,
            const float* __restrict__ xconv, const float* __restrict__ Alog,
            const float* __restrict__ Dv, float* __restrict__ ybuf, int e) {
  __shared__ float sB[32][DST];
  __shared__ float sC[32][DST];
  const int c = blockIdx.x * blockDim.x + threadIdx.x;   // channel 0..383
  const int b = blockIdx.y;
  float An[DST], hst[DST];
  #pragma unroll
  for (int n = 0; n < DST; ++n) {
    An[n]  = -__expf(Alog[(size_t)c * DST + n]);
    hst[n] = 0.f;
  }
  const float Dc = Dv[c];
  const int base = b * LTOK;

  for (int t0 = 0; t0 < LTOK; t0 += 32) {
    __syncthreads();
    for (int j = threadIdx.x; j < 32 * 2 * DST; j += blockDim.x) {
      int tt = j >> 5, q = j & 31;
      float v = xdb[(size_t)(base + t0 + tt) * NDBL + DTR + q];
      if (q < DST) sB[tt][q] = v; else sC[tt][q - DST] = v;
    }
    __syncthreads();
    for (int ti = 0; ti < 32; ++ti) {
      const int t = t0 + ti;
      const float d = dlt[(size_t)(base + t) * DIN + c];
      const int tok = order_tok(e, t);
      const float u = xconv[(size_t)(base + tok) * DIN + c];
      const float du = d * u;
      float acc = 0.f;
      #pragma unroll
      for (int n = 0; n < DST; ++n) {
        float hv = __expf(d * An[n]) * hst[n] + du * sB[ti][n];
        hst[n] = hv;
        acc += hv * sC[ti][n];
      }
      ybuf[(size_t)(base + tok) * DIN + c] = acc + u * Dc;
    }
  }
}

// y2 = y * silu(z)
__global__ void k_gate_mul(const float* __restrict__ xz, const float* __restrict__ ybuf,
                           float* __restrict__ y2) {
  int i = blockIdx.x * blockDim.x + threadIdx.x;
  if (i >= BL * DIN) return;
  int t = i / DIN, c = i - t * DIN;
  float z = xz[(size_t)t * (2 * DIN) + DIN + c];
  y2[i] = ybuf[i] * siluf(z);
}

// ---------------------------------------------------------------------------
extern "C" void kernel_launch(void* const* d_in, const int* in_sizes, int n_in,
                              void* d_out, int out_size, void* d_ws, size_t ws_size,
                              hipStream_t stream) {
  const float* x    = (const float*)d_in[0];
  const float* Wg   = (const float*)d_in[1];
  const float* bg   = (const float*)d_in[2];
  const float* Win  = (const float*)d_in[3];
  const float* Wx   = (const float*)d_in[4];
  const float* Wdt  = (const float*)d_in[5];
  const float* bdt  = (const float*)d_in[6];
  const float* pw1w = (const float*)d_in[7];
  const float* pw1b = (const float*)d_in[8];
  const float* dww  = (const float*)d_in[9];
  const float* pw2w = (const float*)d_in[10];
  const float* Alog = (const float*)d_in[11];
  const float* Dsk  = (const float*)d_in[12];
  const float* Wout = (const float*)d_in[13];
  float* out = (float*)d_out;

  // Workspace carve-up (floats), reused across the 4 sequential expert passes.
  float* ws    = (float*)d_ws;
  float* gates = ws;                           // BL
  int*   eidx  = (int*)(ws + BL);              // BL
  float* xz    = ws + 2 * BL;                  // BL * 768
  float* hmid  = xz    + (size_t)BL * 2 * DIN; // BL * 24
  float* hmid2 = hmid  + (size_t)BL * MIDC;    // BL * 24
  float* xconv = hmid2 + (size_t)BL * MIDC;    // BL * 384
  float* xdb   = xconv + (size_t)BL * DIN;     // BL * 44
  float* dlt   = xdb   + (size_t)BL * NDBL;    // BL * 384
  float* ybuf  = dlt   + (size_t)BL * DIN;     // BL * 384
  float* y2    = ybuf  + (size_t)BL * DIN;     // BL * 384

  k_zero<<<(BL * DM + 255) / 256, 256, 0, stream>>>(out, BL * DM);
  k_router<<<(BL + 255) / 256, 256, 0, stream>>>(x, Wg, bg, gates, eidx);

  for (int e = 0; e < NEXP; ++e) {
    // xz = x @ Win_e : M=4608, N=768, K=192.  Block tile 16 x 256 (TN=4).
    k_wmma_gemm<DM, 4><<<dim3(BL / 16, (2 * DIN) / 256), 128, 0, stream>>>(
        x, Win + (size_t)e * DM * 2 * DIN, xz, 2 * DIN, 2 * DIN,
        nullptr, nullptr, 0);

    k_pw1<<<(BL * MIDC + 255) / 256, 256, 0, stream>>>(
        xz, pw1w + (size_t)e * DIN * MIDC, pw1b + e * MIDC, hmid);
    k_dwconv<<<(BL * MIDC + 255) / 256, 256, 0, stream>>>(
        hmid, dww + (size_t)e * 9 * MIDC, hmid2);
    k_pw2silu<<<(BL * DIN + 255) / 256, 256, 0, stream>>>(
        hmid2, pw2w + (size_t)e * MIDC * DIN, xconv);
    k_xdbl<<<(BL * NDBL + 255) / 256, 256, 0, stream>>>(
        xconv, Wx + (size_t)e * DIN * NDBL, xdb);
    k_delta<<<(BL * DIN + 255) / 256, 256, 0, stream>>>(
        xdb, Wdt + (size_t)e * DTR * DIN, bdt + e * DIN, dlt);

    k_scan<<<dim3(DIN / 128, BSZ), 128, 0, stream>>>(
        dlt, xdb, xconv, Alog + (size_t)e * DIN * DST, Dsk + e * DIN, ybuf, e);

    k_gate_mul<<<(BL * DIN + 255) / 256, 256, 0, stream>>>(xz, ybuf, y2);

    // out += gate_e * (y2 @ Wout_e) : M=4608, N=192, K=384. Block tile
    // 16 x 192 (TN=3), one block column covers the whole N.
    k_wmma_gemm<DIN, 3><<<dim3(BL / 16, 1), 128, 0, stream>>>(
        y2, Wout + (size_t)e * DIN * DM, out, DM, DM, gates, eidx, e);
  }
}
